// GATDecoder_45827301048643
// MI455X (gfx1250) — compile-verified, matrix operand order checked by reading, not run on hardware
//
#include <hip/hip_runtime.h>
#include <hip/hip_bf16.h>
#include <math.h>

// ---------------------------------------------------------------------------
// Problem constants (from the reference)
// ---------------------------------------------------------------------------
#define NNODES 32000
#define BATCHG 16
#define PERG   2000
#define INDIM  512
#define HID    256
#define EMB    256
#define HEADS  8
#define KH     32            // HID / HEADS
#define NCLS   2
#define EINT   512000
#define EALL   1000000
#define NEG_GAT  0.2f
#define NEG_ACT2 0.1f
#define EPS_BN   1e-5f

typedef __attribute__((ext_vector_type(16))) __bf16 v16bf;
typedef __attribute__((ext_vector_type(8)))  float  v8f;
typedef unsigned short u16;

// round-to-nearest-even f32 -> bf16 (bit-level)
__device__ __forceinline__ u16 f2bf(float f) {
    unsigned u = __float_as_uint(f);
    unsigned lsb = (u >> 16) & 1u;
    return (u16)((u + 0x7FFFu + lsb) >> 16);
}

union BfFrag { v16bf v; u16 u[16]; uint4 q[2]; };

// ---------------------------------------------------------------------------
// one-shot f32 -> bf16 buffer conversion
// ---------------------------------------------------------------------------
__global__ void cvt_bf16_kernel(const float* __restrict__ in, u16* __restrict__ out,
                                long long n) {
    long long stride = (long long)gridDim.x * blockDim.x;
    for (long long i = (long long)blockIdx.x * blockDim.x + threadIdx.x; i < n; i += stride)
        out[i] = f2bf(in[i]);
}

// ---------------------------------------------------------------------------
// WMMA GEMM on pre-converted bf16:  C[M,N] (+)= A[M,K] @ W[N, wofs+K]^T (+bias)
// One 16x64 output strip per wave32: 1 A fragment shared across 4 B fragments,
// 4 v_wmma_f32_16x16x32_bf16 per K-step. Fragment layout per ISA 7.12.2:
//   lanes 0-15 hold K kb..kb+7 (elems 0-7) and kb+16..kb+23 (elems 8-15),
//   lanes 16-31 the same with kb += 8. Each half is one 16B global_load_b128.
// ---------------------------------------------------------------------------
__global__ void __launch_bounds__(256)
wmma_gemm_bf16_kernel(const u16* __restrict__ A, int lda,
                      const u16* __restrict__ W, int ldw, int wofs,
                      const float* __restrict__ bias,
                      float* __restrict__ C, int ldc,
                      int M, int N, int K, int accumulate)
{
    const int wpb     = blockDim.x >> 5;
    const int wave    = blockIdx.x * wpb + (threadIdx.x >> 5);
    const int groupsN = N >> 6;                      // 64-wide column groups
    const int nTiles  = (M >> 4) * groupsN;
    if (wave >= nTiles) return;                      // wave-uniform: EXEC all-1s

    const int tm   = wave / groupsN;
    const int tg   = wave % groupsN;
    const int lane = threadIdx.x & 31;
    const int half = lane >> 4;
    const int l15  = lane & 15;

    const u16* __restrict__ arow = A + (size_t)(tm * 16 + l15) * lda;
    const u16* __restrict__ wr0  = W + (size_t)(tg * 64 +  0 + l15) * ldw + wofs;
    const u16* __restrict__ wr1  = W + (size_t)(tg * 64 + 16 + l15) * ldw + wofs;
    const u16* __restrict__ wr2  = W + (size_t)(tg * 64 + 32 + l15) * ldw + wofs;
    const u16* __restrict__ wr3  = W + (size_t)(tg * 64 + 48 + l15) * ldw + wofs;

    v8f acc0 = {}, acc1 = {}, acc2 = {}, acc3 = {};
    for (int k0 = 0; k0 < K; k0 += 32) {
        const int kb = k0 + half * 8;
        if (k0 + 32 < K)                            // global_prefetch of next A tile
            __builtin_prefetch(arow + kb + 32, 0, 3);

        BfFrag fa, f0, f1, f2, f3;
        fa.q[0] = *(const uint4*)(arow + kb);
        fa.q[1] = *(const uint4*)(arow + kb + 16);
        f0.q[0] = *(const uint4*)(wr0 + kb);  f0.q[1] = *(const uint4*)(wr0 + kb + 16);
        f1.q[0] = *(const uint4*)(wr1 + kb);  f1.q[1] = *(const uint4*)(wr1 + kb + 16);
        f2.q[0] = *(const uint4*)(wr2 + kb);  f2.q[1] = *(const uint4*)(wr2 + kb + 16);
        f3.q[0] = *(const uint4*)(wr3 + kb);  f3.q[1] = *(const uint4*)(wr3 + kb + 16);

        acc0 = __builtin_amdgcn_wmma_f32_16x16x32_bf16(false, fa.v, false, f0.v,
                                                       (short)0, acc0, false, false);
        acc1 = __builtin_amdgcn_wmma_f32_16x16x32_bf16(false, fa.v, false, f1.v,
                                                       (short)0, acc1, false, false);
        acc2 = __builtin_amdgcn_wmma_f32_16x16x32_bf16(false, fa.v, false, f2.v,
                                                       (short)0, acc2, false, false);
        acc3 = __builtin_amdgcn_wmma_f32_16x16x32_bf16(false, fa.v, false, f3.v,
                                                       (short)0, acc3, false, false);
    }

    // D layout: elem i -> row = tm*16 + half*8 + i, col = colbase + (lane&15)
    const int rbase = tm * 16 + half * 8;
    v8f accs[4] = {acc0, acc1, acc2, acc3};
#pragma unroll
    for (int s = 0; s < 4; ++s) {
        const int col = tg * 64 + s * 16 + l15;
        const float bc = bias ? bias[col] : 0.0f;
#pragma unroll
        for (int i = 0; i < 8; ++i) {
            size_t idx = (size_t)(rbase + i) * ldc + col;
            float v = accs[s][i] + bc;
            if (accumulate) v += C[idx];
            C[idx] = v;
        }
    }
}

// ---------------------------------------------------------------------------
// Elementwise / edge kernels (f32 scalar path)
// ---------------------------------------------------------------------------
__global__ void fill_kernel(float* __restrict__ p, float v, long long n) {
    long long stride = (long long)gridDim.x * blockDim.x;
    for (long long i = (long long)blockIdx.x * blockDim.x + threadIdx.x; i < n; i += stride)
        p[i] = v;
}

__device__ __forceinline__ void atomicMaxF(float* addr, float val) {
    int old = __float_as_int(*addr);
    while (__int_as_float(old) < val) {
        int assumed = old;
        old = atomicCAS((int*)addr, assumed, __float_as_int(val));
        if (old == assumed) break;
    }
}

// per-node attention scalars: sdst = h . a[:, :k], ssrc = h . a[:, k:]
__global__ void node_att_kernel(const float* __restrict__ h,
                                const float* __restrict__ att,
                                float* __restrict__ sdst,
                                float* __restrict__ ssrc) {
    int idx = blockIdx.x * blockDim.x + threadIdx.x;
    if (idx >= NNODES * HEADS) return;
    int node = idx / HEADS, hd = idx % HEADS;
    const float* hp = h + (size_t)node * HID + hd * KH;
    const float* ad = att + hd * 2 * KH;
    float s0 = 0.f, s1 = 0.f;
#pragma unroll
    for (int k = 0; k < KH; ++k) { float v = hp[k]; s0 += v * ad[k]; s1 += v * ad[KH + k]; }
    sdst[idx] = s0; ssrc[idx] = s1;
}

__device__ __forceinline__ void edge_pair(const int* __restrict__ ei, int E, int e,
                                          int& src, int& dst) {
    if (e < E) { src = ei[e]; dst = ei[E + e]; }
    else       { src = dst = e - E; }                // self loops
}

__global__ void edge_alpha_kernel(const int* __restrict__ ei, int E, int etot,
                                  const float* __restrict__ sdst,
                                  const float* __restrict__ ssrc,
                                  float* __restrict__ alpha,
                                  float* __restrict__ maxb) {
    long long total = (long long)etot * HEADS;
    long long stride = (long long)gridDim.x * blockDim.x;
    for (long long t = (long long)blockIdx.x * blockDim.x + threadIdx.x; t < total; t += stride) {
        int e = (int)(t / HEADS), hd = (int)(t % HEADS);
        int src, dst; edge_pair(ei, E, e, src, dst);
        float a = sdst[dst * HEADS + hd] + ssrc[src * HEADS + hd];
        a = a > 0.f ? a : NEG_GAT * a;               // leaky_relu(0.2)
        alpha[t] = a;
        atomicMaxF(&maxb[src * HEADS + hd], a);      // segment max over src
    }
}

__global__ void edge_exp_kernel(const int* __restrict__ ei, int E, int etot,
                                float* __restrict__ alpha,
                                const float* __restrict__ maxb,
                                float* __restrict__ denom) {
    long long total = (long long)etot * HEADS;
    long long stride = (long long)gridDim.x * blockDim.x;
    for (long long t = (long long)blockIdx.x * blockDim.x + threadIdx.x; t < total; t += stride) {
        int e = (int)(t / HEADS), hd = (int)(t % HEADS);
        int src, dst; edge_pair(ei, E, e, src, dst); (void)dst;
        float ev = expf(alpha[t] - maxb[src * HEADS + hd]);
        alpha[t] = ev;
        atomicAdd(&denom[src * HEADS + hd], ev);     // segment sum over src
    }
}

// 4 features per thread: float4 gather of source row, 4 scatter atomics
__global__ void edge_scatter_kernel(const int* __restrict__ ei, int E, int etot,
                                    const float* __restrict__ h,
                                    const float* __restrict__ alpha,
                                    const float* __restrict__ denom,
                                    float* __restrict__ aggr) {
    const int F4 = HID / 4;                          // 64 float4 chunks per row
    long long total = (long long)etot * F4;
    long long stride = (long long)gridDim.x * blockDim.x;
    for (long long t = (long long)blockIdx.x * blockDim.x + threadIdx.x; t < total; t += stride) {
        int e = (int)(t / F4), c = (int)(t % F4);
        int f = c * 4, hd = f >> 5;                  // f / KH
        int src, dst; edge_pair(ei, E, e, src, dst);
        float w = alpha[(long long)e * HEADS + hd] /
                  (denom[src * HEADS + hd] + 1e-16f);
        float4 hv = *(const float4*)(h + (size_t)src * HID + f);
        float* out = aggr + (size_t)dst * HID + f;
        atomicAdd(out + 0, hv.x * w);
        atomicAdd(out + 1, hv.y * w);
        atomicAdd(out + 2, hv.z * w);
        atomicAdd(out + 3, hv.w * w);
    }
}

// y = relu(aggr + bias); fused column sum / sumsq for BatchNorm
__global__ void gat_finalize_kernel(float* __restrict__ aggr,
                                    const float* __restrict__ bias,
                                    float* __restrict__ sum,
                                    float* __restrict__ sq) {
    long long total = (long long)NNODES * HID;
    long long stride = (long long)gridDim.x * blockDim.x;
    for (long long i = (long long)blockIdx.x * blockDim.x + threadIdx.x; i < total; i += stride) {
        int f = (int)(i % HID);
        float v = aggr[i] + bias[f];
        v = v > 0.f ? v : 0.f;
        aggr[i] = v;
        atomicAdd(&sum[f], v);
        atomicAdd(&sq[f], v * v);
    }
}

// out = act( g * (y - mean) * rsqrt(var + eps) + b ), biased batch stats
__global__ void bn_act_kernel(const float* __restrict__ y,
                              const float* __restrict__ sum,
                              const float* __restrict__ sq,
                              const float* __restrict__ g,
                              const float* __restrict__ bt,
                              float* __restrict__ out, float negslope) {
    long long total = (long long)NNODES * HID;
    long long stride = (long long)gridDim.x * blockDim.x;
    const float invn = 1.0f / (float)NNODES;
    for (long long i = (long long)blockIdx.x * blockDim.x + threadIdx.x; i < total; i += stride) {
        int f = (int)(i % HID);
        float mean = sum[f] * invn;
        float var  = sq[f] * invn - mean * mean;
        float v = g[f] * (y[i] - mean) * rsqrtf(var + EPS_BN) + bt[f];
        out[i] = v > 0.f ? v : negslope * v;
    }
}

// PowerMean pooling: clip, pow(p), mean over nodes, clip, pow(1/p)
__global__ void pool_kernel(const float* __restrict__ xg,
                            const float* __restrict__ p_ptr,
                            float* __restrict__ pooled) {
    int idx = blockIdx.x * blockDim.x + threadIdx.x;
    if (idx >= BATCHG * EMB) return;
    int b = idx / EMB, f = idx % EMB;
    float p = p_ptr[0];
    float s = 0.f;
    for (int n = 0; n < PERG; ++n) {
        float v = xg[((size_t)b * PERG + n) * EMB + f];
        v = fminf(fmaxf(v, 1e-6f), 100.f);
        s += powf(v, p);
    }
    s *= (1.0f / (float)PERG);
    s = fminf(fmaxf(s, 1e-6f), 100.f);
    pooled[idx] = powf(s, 1.0f / p);
}

// logits = pooled @ w_pred^T + b_pred ; ypred = argmax
__global__ void head_kernel(const float* __restrict__ pooled,
                            const float* __restrict__ wp,
                            const float* __restrict__ bp,
                            float* __restrict__ out) {
    int b = threadIdx.x;
    if (b >= BATCHG) return;
    float o0 = bp[0], o1 = bp[1];
    for (int k = 0; k < EMB; ++k) {
        float v = pooled[b * EMB + k];
        o0 += v * wp[k];
        o1 += v * wp[EMB + k];
    }
    out[b * NCLS + 0] = o0;
    out[b * NCLS + 1] = o1;
    out[BATCHG * NCLS + b] = (o1 > o0) ? 1.0f : 0.0f;   // argmax, ties -> 0
}

// ---------------------------------------------------------------------------
// Host-side orchestration
// ---------------------------------------------------------------------------
static void launch_cvt(hipStream_t s, const float* in, u16* out, long long n) {
    int blocks = (int)((n + 1023) / 1024);
    if (blocks > 4096) blocks = 4096;
    if (blocks < 1)    blocks = 1;
    cvt_bf16_kernel<<<blocks, 256, 0, s>>>(in, out, n);
}

static void launch_gemm(hipStream_t s, const u16* A, int lda,
                        const u16* W, int ldw, int wofs, const float* bias,
                        float* C, int ldc, int M, int N, int K, int acc) {
    int tiles = (M / 16) * (N / 64);                 // 16x64 strip per wave
    int blocks = (tiles + 7) / 8;                    // 8 waves / 256-thread block
    wmma_gemm_bf16_kernel<<<blocks, 256, 0, s>>>(A, lda, W, ldw, wofs, bias, C, ldc,
                                                 M, N, K, acc);
}

struct GatParams { const u16* wbf; const float *a, *b, *g, *bt; };

static void run_gat(hipStream_t s, const float* xin, const GatParams& P,
                    const int* ei, int E, float negslope,
                    u16* abf, float* hbuf, float* alphab, float* sdst, float* ssrc,
                    float* maxb, float* denom, float* aggr,
                    float* sumb, float* sqb, float* out) {
    const int etot = E + NNODES;
    const long long NH = (long long)NNODES * HID;

    // projection h = xin @ W^T via WMMA (activations converted to bf16 once)
    launch_cvt(s, xin, abf, NH);
    launch_gemm(s, abf, HID, P.wbf, HID, 0, nullptr, hbuf, HID, NNODES, HID, HID, 0);

    // reset scratch (deterministic under graph replay)
    fill_kernel<<<2048, 256, 0, s>>>(aggr, 0.f, NH);
    fill_kernel<<<256, 256, 0, s>>>(maxb, -1e30f, (long long)NNODES * HEADS);
    fill_kernel<<<256, 256, 0, s>>>(denom, 0.f, (long long)NNODES * HEADS);
    fill_kernel<<<1, 256, 0, s>>>(sumb, 0.f, HID);
    fill_kernel<<<1, 256, 0, s>>>(sqb, 0.f, HID);

    node_att_kernel<<<(NNODES * HEADS + 255) / 256, 256, 0, s>>>(hbuf, P.a, sdst, ssrc);
    edge_alpha_kernel<<<4096, 256, 0, s>>>(ei, E, etot, sdst, ssrc, alphab, maxb);
    edge_exp_kernel<<<4096, 256, 0, s>>>(ei, E, etot, alphab, maxb, denom);
    edge_scatter_kernel<<<8192, 256, 0, s>>>(ei, E, etot, hbuf, alphab, denom, aggr);
    gat_finalize_kernel<<<4096, 256, 0, s>>>(aggr, P.b, sumb, sqb);
    bn_act_kernel<<<4096, 256, 0, s>>>(aggr, sumb, sqb, P.g, P.bt, out, negslope);
}

extern "C" void kernel_launch(void* const* d_in, const int* in_sizes, int n_in,
                              void* d_out, int out_size, void* d_ws, size_t ws_size,
                              hipStream_t stream) {
    (void)in_sizes; (void)n_in; (void)out_size; (void)ws_size;

    const float* x      = (const float*)d_in[0];
    const int*   ei_int = (const int*)  d_in[1];
    const int*   ei_all = (const int*)  d_in[2];
    const float* w_it   = (const float*)d_in[3];
    const float* b_it   = (const float*)d_in[4];
    const float* w_m    = (const float*)d_in[5];
    const float* b_m    = (const float*)d_in[6];
    const float* w_pred = (const float*)d_in[7];
    const float* b_pred = (const float*)d_in[8];
    const float* p      = (const float*)d_in[9];

    // workspace bump allocator
    char* w = (char*)d_ws;
    auto allocf = [&](size_t n) { float* q = (float*)w; w += n * sizeof(float); return q; };
    auto allocb = [&](size_t n) { u16*   q = (u16*)w;   w += ((n + 1) & ~1ull) * sizeof(u16); return q; };

    float* f0     = allocf((size_t)NNODES * HID);
    float* f1     = allocf((size_t)NNODES * HID);
    float* hbuf   = allocf((size_t)NNODES * HID);
    float* aggr   = allocf((size_t)NNODES * HID);
    float* sdst   = allocf((size_t)NNODES * HEADS);
    float* ssrc   = allocf((size_t)NNODES * HEADS);
    float* maxb   = allocf((size_t)NNODES * HEADS);
    float* denom  = allocf((size_t)NNODES * HEADS);
    float* alphab = allocf((size_t)(EALL + NNODES) * HEADS);
    float* sumb   = allocf(HID);
    float* sqb    = allocf(HID);
    float* pooled = allocf((size_t)BATCHG * EMB);
    u16*   xbf    = allocb((size_t)NNODES * INDIM);      // x in bf16
    u16*   abf    = allocb((size_t)NNODES * HID);        // per-layer activations bf16
    u16*   wbf_it = allocb((size_t)HID * INDIM);
    u16*   wbf_m  = allocb((size_t)HID * (INDIM + HID));
    u16*   wbf_l[6];
    for (int i = 0; i < 6; ++i) wbf_l[i] = allocb((size_t)HID * HID);

    GatParams ig[3], gg[3];
    for (int i = 0; i < 3; ++i) {
        ig[i] = { wbf_l[i],     (const float*)d_in[11 + 5 * i], (const float*)d_in[12 + 5 * i],
                                (const float*)d_in[13 + 5 * i], (const float*)d_in[14 + 5 * i] };
        gg[i] = { wbf_l[3 + i], (const float*)d_in[26 + 5 * i], (const float*)d_in[27 + 5 * i],
                                (const float*)d_in[28 + 5 * i], (const float*)d_in[29 + 5 * i] };
    }

    // 0) one-shot bf16 conversions of x and all GEMM weights
    launch_cvt(stream, x,    xbf,    (long long)NNODES * INDIM);
    launch_cvt(stream, w_it, wbf_it, (long long)HID * INDIM);
    launch_cvt(stream, w_m,  wbf_m,  (long long)HID * (INDIM + HID));
    for (int i = 0; i < 3; ++i) {
        launch_cvt(stream, (const float*)d_in[10 + 5 * i], wbf_l[i],     (long long)HID * HID);
        launch_cvt(stream, (const float*)d_in[25 + 5 * i], wbf_l[3 + i], (long long)HID * HID);
    }

    // 1) xi = x @ w_it^T + b_it
    launch_gemm(stream, xbf, INDIM, wbf_it, INDIM, 0, b_it, f0, HID, NNODES, HID, INDIM, 0);

    // 2) three internal GAT layers (BN + relu), ping-pong f0 <-> f1
    float* cur = f0; float* nxt = f1;
    for (int i = 0; i < 3; ++i) {
        run_gat(stream, cur, ig[i], ei_int, EINT, 0.0f,
                abf, hbuf, alphab, sdst, ssrc, maxb, denom, aggr, sumb, sqb, nxt);
        float* t = cur; cur = nxt; nxt = t;
    }
    // internal branch output now in cur (== f1)

    // 3) merge: xc = x@Wm[:, :512]^T + xi@Wm[:, 512:]^T + b_m
    launch_cvt(stream, cur, abf, (long long)NNODES * HID);
    launch_gemm(stream, xbf, INDIM, wbf_m, INDIM + HID, 0,   b_m,     f0, HID, NNODES, HID, INDIM, 0);
    launch_gemm(stream, abf, HID,   wbf_m, INDIM + HID, 512, nullptr, f0, HID, NNODES, HID, HID,   1);

    // 4) three global GAT layers (BN + leaky 0.1), ping-pong again
    cur = f0; nxt = f1;
    for (int i = 0; i < 3; ++i) {
        run_gat(stream, cur, gg[i], ei_all, EALL, NEG_ACT2,
                abf, hbuf, alphab, sdst, ssrc, maxb, denom, aggr, sumb, sqb, nxt);
        float* t = cur; cur = nxt; nxt = t;
    }
    // final node embeddings in cur (== f1)

    // 5) power-mean pooling + prediction head (+ argmax)
    pool_kernel<<<(BATCHG * EMB + 255) / 256, 256, 0, stream>>>(cur, p, pooled);
    head_kernel<<<1, 32, 0, stream>>>(pooled, w_pred, b_pred, (float*)d_out);
}
